// VideoChapterWindowAttention_9706626089218
// MI455X (gfx1250) — matrix operand order repaired and checked by reading, not compiled, not run on hardware
//
#include <hip/hip_runtime.h>

// ---------------------------------------------------------------------------
// MI455X (gfx1250, wave32) attention: bf16 WMMA + TDM (tensor_load_to_lds).
// All LDS fragment reads are k-contiguous -> ds_load_b128.
// B=16, S=1024, HID=1024, H=16, D=64, pos_bias window = 1025 (use first 1024).
// ---------------------------------------------------------------------------

#define B_   16
#define S_   1024
#define HID_ 1024
#define H_   16
#define D_   64

typedef __attribute__((ext_vector_type(16))) __bf16 v16bf;
typedef __attribute__((ext_vector_type(8)))  float  v8f;
typedef __attribute__((ext_vector_type(4)))  unsigned int v4u;
typedef __attribute__((ext_vector_type(8)))  int v8i;
typedef __attribute__((ext_vector_type(4)))  int v4i;

union FragB16 {
    v16bf v;
    unsigned int   u[8];
    unsigned short s[16];
};

__device__ __forceinline__ unsigned short f32_to_bf16(float f) {
    unsigned int u = __float_as_uint(f);
    u += 0x7FFFu + ((u >> 16) & 1u);       // round-to-nearest-even
    return (unsigned short)(u >> 16);
}

__device__ __forceinline__ v8f vzero() {
    v8f z = {0.f, 0.f, 0.f, 0.f, 0.f, 0.f, 0.f, 0.f};
    return z;
}

__device__ __forceinline__ v8f wmma_bf16(const FragB16& a, const FragB16& b, v8f c) {
    return __builtin_amdgcn_wmma_f32_16x16x32_bf16(
        /*neg_a=*/false, a.v, /*neg_b=*/false, b.v,
        /*c_mod=*/(short)0, c, /*reuse_a=*/false, /*reuse_b=*/false);
}

__device__ __forceinline__ unsigned lds_off(const void* p) {
    return (unsigned)(unsigned long long)p;   // low 32 bits = LDS byte offset
}

// ---------------------------------------------------------------------------
// TDM: load a 2-D tile (tile_d1 rows x tile_d0 bf16 elements, row stride
// stride0 elements) from global memory into LDS at lds_addr (packed rows).
// Builds D# group0/group1 per CDNA5 ISA ch.8 (data_size=1 -> 2 bytes).
// ---------------------------------------------------------------------------
__device__ __forceinline__ void tdm_load_2d(unsigned lds_addr, const void* gptr,
                                            unsigned tensor_d0, unsigned tensor_d1,
                                            unsigned tile_d0, unsigned tile_d1,
                                            unsigned stride0) {
    unsigned long long ga = (unsigned long long)gptr;
    v4u g0;
    g0[0] = 1u;                                            // count=1, user mode
    g0[1] = lds_addr;                                      // lds_addr [63:32]
    g0[2] = (unsigned)(ga & 0xFFFFFFFFu);                  // global_addr lo
    g0[3] = (unsigned)(((ga >> 32) & 0x01FFFFFFu)          // global_addr hi (57b)
                       | 0x80000000u);                     // type=2 ("image")
    v8i g1;
    g1[0] = (int)(1u << 16);                               // data_size=1 (2B), mask=0
    g1[1] = (int)((tensor_d0 & 0xFFFFu) << 16);            // tensor_dim0 [63:48]
    g1[2] = (int)(((tensor_d0 >> 16) & 0xFFFFu)            // tensor_dim0 [79:64]
                  | ((tensor_d1 & 0xFFFFu) << 16));        // tensor_dim1 [95:80]
    g1[3] = (int)(((tensor_d1 >> 16) & 0xFFFFu)            // tensor_dim1 [111:96]
                  | ((tile_d0 & 0xFFFFu) << 16));          // tile_dim0  [127:112]
    g1[4] = (int)(tile_d1 & 0xFFFFu);                      // tile_dim1, tile_dim2=0
    g1[5] = (int)stride0;                                  // tensor_dim0_stride lo
    g1[6] = 0;                                             // stride hi, dim1_stride lo
    g1[7] = 0;                                             // dim1_stride hi
    v4i z = {0, 0, 0, 0};
#if defined(__clang_major__) && (__clang_major__ >= 23)
    v8i z8 = {0, 0, 0, 0, 0, 0, 0, 0};
    __builtin_amdgcn_tensor_load_to_lds(g0, g1, z, z, z8, 0);
#else
    __builtin_amdgcn_tensor_load_to_lds(g0, g1, z, z, 0);
#endif
}

// A-fragment (16x32 bf16, MxK): element (m,k) at base[m*ld + k].
__device__ __forceinline__ void load_a_frag(FragB16& f, const unsigned short* base,
                                            int ld, int lane) {
    const int m = lane & 15, h = lane >> 4;
    const unsigned short* p = base + m * ld + h * 8;
#pragma unroll
    for (int v = 0; v < 8; ++v) {
        const int k = ((v & 4) ? 16 : 0) + (v & 3) * 2;
        f.u[v] = *(const unsigned int*)(p + k);
    }
}

// B-fragment (32x16 bf16, KxN): element (k,n) at base[n*ld + k] (k contiguous).
__device__ __forceinline__ void load_b_frag_kc(FragB16& f, const unsigned short* base,
                                               int ld, int lane) {
    const int n = lane & 15, h = lane >> 4;
    const unsigned short* p = base + n * ld + h * 8;
#pragma unroll
    for (int v = 0; v < 8; ++v)
        f.u[v] = *(const unsigned int*)(p + ((v & 4) ? 16 : 0) + (v & 3) * 2);
}

// ---------------------------------------------------------------------------
// Kernel 1: hs = hidden + positions*W_pe^T + b_pe, convert to bf16.
// ---------------------------------------------------------------------------
__global__ __launch_bounds__(256)
void prep_hs_kernel(const float* __restrict__ hidden,
                    const float* __restrict__ Wpe,
                    const float* __restrict__ bpe,
                    unsigned short* __restrict__ hsb) {
    const int idx = blockIdx.x * 256 + threadIdx.x;       // over 16384*1024
    const int n = idx & (HID_ - 1);
    const int m = idx >> 10;
    const int s = m & (S_ - 1);
    const float pos = (float)(s - S_ / 2) / (512.0f + 1e-6f);
    hsb[idx] = f32_to_bf16(hidden[idx] + pos * Wpe[n] + bpe[n]);
}

// ---------------------------------------------------------------------------
// Kernel 2: straight fp32 -> bf16 weight convert (keeps [N,K] layout).
// ---------------------------------------------------------------------------
__global__ __launch_bounds__(256)
void wconv_kernel(const float* __restrict__ W, unsigned short* __restrict__ Wb) {
    const int idx = blockIdx.x * 256 + threadIdx.x;       // over 1024*1024
    Wb[idx] = f32_to_bf16(W[idx]);
}

// ---------------------------------------------------------------------------
// Kernel 3/5: C[m,n] = sum_k A[m,k] * Wb[n,k] + bias[n]
// MODE 0: bf16 out scattered to [B,H,S,D] (Q,K).
// MODE 1: fp32 out, plain [M,N] (output projection).
// MODE 2: bf16 out scattered to [B,H,D,S] (V transposed, for attention TDM).
// 128x128 block, 256 threads = 8 waves (2x4); each wave 64x32 = 4x2 WMMA tiles.
// TDM double-buffered LDS tiles, K-step 32; all frag reads k-contiguous.
// ---------------------------------------------------------------------------
template <int MODE>
__global__ __launch_bounds__(256)
void gemm_bf16_wmma(const unsigned short* __restrict__ A,
                    const unsigned short* __restrict__ Wb,
                    const float* __restrict__ bias,
                    unsigned short* __restrict__ obf,
                    float* __restrict__ of32,
                    int M, int N, int K) {
    __shared__ unsigned short As[2][128][32];   // [m][k], TDM packed rows
    __shared__ unsigned short Bs[2][128][32];   // [n][k], TDM packed rows

    const int t = threadIdx.x;
    const int lane = t & 31;
    const int wave = t >> 5;
    const int wm = wave >> 2;                  // 0..1
    const int wn = wave & 3;                   // 0..3
    const int brow = blockIdx.y * 128;
    const int bcol = blockIdx.x * 128;

    auto issue = [&](int kt, int buf) {
        // A tile: 128 m-rows x 32 k, row stride K
        tdm_load_2d(lds_off(&As[buf][0][0]),
                    A + (size_t)brow * K + (size_t)kt * 32,
                    32, 128, 32, 128, (unsigned)K);
        // B tile: 128 n-rows x 32 k, row stride K (weights kept [N,K])
        tdm_load_2d(lds_off(&Bs[buf][0][0]),
                    Wb + (size_t)bcol * K + (size_t)kt * 32,
                    32, 128, 32, 128, (unsigned)K);
    };

    v8f acc[4][2];
#pragma unroll
    for (int mt = 0; mt < 4; ++mt)
#pragma unroll
        for (int nt = 0; nt < 2; ++nt) acc[mt][nt] = vzero();

    const int NK = K / 32;
    if (wave == 0) issue(0, 0);

    for (int kt = 0; kt < NK; ++kt) {
        const int cur = kt & 1;
        const bool more = (kt + 1) < NK;
        if (wave == 0) {
            if (more) {
                issue(kt + 1, cur ^ 1);
                __builtin_amdgcn_s_wait_tensorcnt((short)2);  // tile kt complete
            } else {
                __builtin_amdgcn_s_wait_tensorcnt((short)0);
            }
        }
        __syncthreads();

        FragB16 af[4], bfr[2];
#pragma unroll
        for (int mt = 0; mt < 4; ++mt)
            load_a_frag(af[mt], &As[cur][wm * 64 + mt * 16][0], 32, lane);
#pragma unroll
        for (int nt = 0; nt < 2; ++nt)
            load_b_frag_kc(bfr[nt], &Bs[cur][wn * 32 + nt * 16][0], 32, lane);
#pragma unroll
        for (int mt = 0; mt < 4; ++mt)
#pragma unroll
            for (int nt = 0; nt < 2; ++nt)
                acc[mt][nt] = wmma_bf16(af[mt], bfr[nt], acc[mt][nt]);

        __syncthreads();
    }

    const int hh = lane >> 4, nn = lane & 15;
#pragma unroll
    for (int mt = 0; mt < 4; ++mt)
#pragma unroll
        for (int nt = 0; nt < 2; ++nt) {
            const int gm = brow + wm * 64 + mt * 16;
            const int gn = bcol + wn * 32 + nt * 16 + nn;
            const float bv = bias[gn];
#pragma unroll
            for (int r = 0; r < 8; ++r) {
                const int m = gm + r + 8 * hh;
                const float val = acc[mt][nt][r] + bv;
                if (MODE == 0) {
                    // scatter into [B, H, S, D]
                    const int bb = m >> 10, s = m & (S_ - 1);
                    const int hd = gn >> 6, d = gn & (D_ - 1);
                    obf[((size_t)(bb * H_ + hd) * S_ + s) * D_ + d] = f32_to_bf16(val);
                } else if (MODE == 2) {
                    // scatter into [B, H, D, S]  (V transposed)
                    const int bb = m >> 10, s = m & (S_ - 1);
                    const int hd = gn >> 6, d = gn & (D_ - 1);
                    obf[((size_t)(bb * H_ + hd) * D_ + d) * S_ + s] = f32_to_bf16(val);
                } else {
                    of32[(size_t)m * N + gn] = val;
                }
            }
        }
}

// ---------------------------------------------------------------------------
// Kernel 4: flash attention per (b, h, 128-query block).
// 8 waves: wave w owns query rows [w*16, w*16+16). Online softmax per wave.
// K tile [128 key][64 d] (contiguous), V tile [64 d][128 key] (stride S) via TDM.
// ---------------------------------------------------------------------------
__global__ __launch_bounds__(256)
void attn_wmma(const unsigned short* __restrict__ Q,
               const unsigned short* __restrict__ Km,
               const unsigned short* __restrict__ Vt,
               const float* __restrict__ pos_bias,
               unsigned short* __restrict__ ctx) {
    __shared__ unsigned short Ks[128][64];       // 16 KB  (key, d)
    __shared__ unsigned short Vts[64][128];      // 16 KB  (d, key)
    __shared__ unsigned short Ps[8][16][128];    // 32 KB  per-wave probs

    const int t = threadIdx.x, lane = t & 31, wave = t >> 5;
    const int h = blockIdx.y, b = blockIdx.z;
    const int q0 = blockIdx.x * 128 + wave * 16;
    const size_t headoff = (size_t)(b * H_ + h) * S_ * D_;   // same size both layouts

    // Q A-fragments for k-steps 0/1 of D=64 (resident whole kernel)
    FragB16 aq[2];
    {
        const int m = lane & 15, hh0 = lane >> 4;
        const unsigned short* qrow = Q + headoff + (size_t)(q0 + m) * D_ + hh0 * 8;
#pragma unroll
        for (int ks = 0; ks < 2; ++ks)
#pragma unroll
            for (int v = 0; v < 8; ++v) {
                const int k = ks * 32 + ((v & 4) ? 16 : 0) + (v & 3) * 2;
                aq[ks].u[v] = *(const unsigned int*)(qrow + k);
            }
    }

    float mrow[8], lrow[8];
    v8f cacc[4];
#pragma unroll
    for (int r = 0; r < 8; ++r) { mrow[r] = -3.0e38f; lrow[r] = 0.f; }
#pragma unroll
    for (int dt = 0; dt < 4; ++dt) cacc[dt] = vzero();

    const int hh = lane >> 4, nn = lane & 15;

    for (int kb = 0; kb < 8; ++kb) {
        const int key0 = kb * 128;
        __syncthreads();                      // prior Ks/Vts reads done
        if (wave == 0) {
            // K tile: 128x64 contiguous bf16 = 8192 elements
            tdm_load_2d(lds_off(&Ks[0][0]),
                        Km + headoff + (size_t)key0 * D_, 8192, 1, 8192, 1, 8192);
            // V tile: 64 d-rows x 128 keys, row stride S
            tdm_load_2d(lds_off(&Vts[0][0]),
                        Vt + headoff + (size_t)key0, 128, 64, 128, 64, S_);
            __builtin_amdgcn_s_wait_tensorcnt((short)0);
        }
        if (kb + 1 < 8)
            __builtin_prefetch(pos_bias + h * (2 * 512 + 1) + (kb + 1) * 128, 0, 3);
        __syncthreads();

        // scores tile 16x128: S = Q @ K^T
        v8f sacc[8];
#pragma unroll
        for (int nt = 0; nt < 8; ++nt) sacc[nt] = vzero();
#pragma unroll
        for (int nt = 0; nt < 8; ++nt) {
            FragB16 bk;
#pragma unroll
            for (int ks = 0; ks < 2; ++ks) {
                load_b_frag_kc(bk, &Ks[nt * 16][ks * 32], 64, lane);
                sacc[nt] = wmma_bf16(aq[ks], bk, sacc[nt]);
            }
        }

        // scale + positional bias + row max
        float tmax[8];
#pragma unroll
        for (int r = 0; r < 8; ++r) tmax[r] = -3.0e38f;
#pragma unroll
        for (int nt = 0; nt < 8; ++nt) {
            const float pb = pos_bias[h * (2 * 512 + 1) + key0 + nt * 16 + nn];
#pragma unroll
            for (int r = 0; r < 8; ++r) {
                const float x = sacc[nt][r] * 0.125f + pb;
                sacc[nt][r] = x;
                tmax[r] = fmaxf(tmax[r], x);
            }
        }
#pragma unroll
        for (int mk = 1; mk <= 8; mk <<= 1)
#pragma unroll
            for (int r = 0; r < 8; ++r)
                tmax[r] = fmaxf(tmax[r], __shfl_xor(tmax[r], mk, 32));

        // online softmax update
        float scale[8], lsum[8];
#pragma unroll
        for (int r = 0; r < 8; ++r) {
            const float mn = fmaxf(mrow[r], tmax[r]);
            scale[r] = __expf(mrow[r] - mn);
            mrow[r] = mn;
            lsum[r] = 0.f;
        }
#pragma unroll
        for (int nt = 0; nt < 8; ++nt)
#pragma unroll
            for (int r = 0; r < 8; ++r) {
                const float p = __expf(sacc[nt][r] - mrow[r]);
                lsum[r] += p;
                Ps[wave][r + 8 * hh][nt * 16 + nn] = f32_to_bf16(p);
            }
#pragma unroll
        for (int mk = 1; mk <= 8; mk <<= 1)
#pragma unroll
            for (int r = 0; r < 8; ++r)
                lsum[r] += __shfl_xor(lsum[r], mk, 32);
#pragma unroll
        for (int r = 0; r < 8; ++r)
            lrow[r] = lrow[r] * scale[r] + lsum[r];
#pragma unroll
        for (int dt = 0; dt < 4; ++dt)
#pragma unroll
            for (int r = 0; r < 8; ++r)
                cacc[dt][r] *= scale[r];

        // ctx += P(16x128) @ V(128x64), V read from transposed LDS tile
#pragma unroll
        for (int dt = 0; dt < 4; ++dt)
#pragma unroll
            for (int kk = 0; kk < 4; ++kk) {
                FragB16 ap, bv;
                load_a_frag(ap, &Ps[wave][0][kk * 32], 128, lane);
                load_b_frag_kc(bv, &Vts[dt * 16][kk * 32], 128, lane);
                cacc[dt] = wmma_bf16(ap, bv, cacc[dt]);
            }
    }

    // normalize and scatter to [B, S, H, D] (GEMM-A layout for out-proj)
    float inv[8];
#pragma unroll
    for (int r = 0; r < 8; ++r) inv[r] = 1.0f / lrow[r];
#pragma unroll
    for (int dt = 0; dt < 4; ++dt)
#pragma unroll
        for (int r = 0; r < 8; ++r) {
            const int srow = q0 + r + 8 * hh;
            const int d = dt * 16 + nn;
            ctx[((size_t)(b * S_ + srow) * H_ + h) * D_ + d] =
                f32_to_bf16(cacc[dt][r] * inv[r]);
        }
}

// ---------------------------------------------------------------------------
// Launcher
// ---------------------------------------------------------------------------
extern "C" void kernel_launch(void* const* d_in, const int* in_sizes, int n_in,
                              void* d_out, int out_size, void* d_ws, size_t ws_size,
                              hipStream_t stream) {
    const float* hidden   = (const float*)d_in[0];
    const float* Wq       = (const float*)d_in[1];
    const float* bq       = (const float*)d_in[2];
    const float* Wk       = (const float*)d_in[3];
    const float* bk       = (const float*)d_in[4];
    const float* Wv       = (const float*)d_in[5];
    const float* bv       = (const float*)d_in[6];
    const float* Wo       = (const float*)d_in[7];
    const float* bo       = (const float*)d_in[8];
    const float* Wpe      = (const float*)d_in[9];
    const float* bpe      = (const float*)d_in[10];
    const float* pos_bias = (const float*)d_in[11];
    float* out = (float*)d_out;

    char* ws = (char*)d_ws;
    size_t off = 0;
    auto walloc = [&](size_t bytes) -> void* {
        void* p = ws + off;
        off += (bytes + 255) & ~(size_t)255;
        return p;
    };
    const size_t MKbytes = (size_t)B_ * S_ * HID_ * 2;       // 32 MB (bf16 [M,K])
    const size_t Wbytes  = (size_t)HID_ * HID_ * 2;          // 2 MB
    const size_t QKVb    = (size_t)B_ * H_ * S_ * D_ * 2;    // 32 MB

    unsigned short* hsb  = (unsigned short*)walloc(MKbytes);
    unsigned short* Wqb  = (unsigned short*)walloc(Wbytes);
    unsigned short* Wkb  = (unsigned short*)walloc(Wbytes);
    unsigned short* Wvb  = (unsigned short*)walloc(Wbytes);
    unsigned short* Wob  = (unsigned short*)walloc(Wbytes);
    unsigned short* qb   = (unsigned short*)walloc(QKVb);
    unsigned short* kbuf = (unsigned short*)walloc(QKVb);
    unsigned short* vbuf = (unsigned short*)walloc(QKVb);   // [B,H,D,S]
    unsigned short* ctx  = (unsigned short*)walloc(MKbytes);

    const int M = B_ * S_, N = HID_, K = HID_;

    prep_hs_kernel<<<(M * HID_) / 256, 256, 0, stream>>>(hidden, Wpe, bpe, hsb);
    wconv_kernel<<<(HID_ * HID_) / 256, 256, 0, stream>>>(Wq, Wqb);
    wconv_kernel<<<(HID_ * HID_) / 256, 256, 0, stream>>>(Wk, Wkb);
    wconv_kernel<<<(HID_ * HID_) / 256, 256, 0, stream>>>(Wv, Wvb);
    wconv_kernel<<<(HID_ * HID_) / 256, 256, 0, stream>>>(Wo, Wob);

    dim3 gg(N / 128, M / 128);
    gemm_bf16_wmma<0><<<gg, 256, 0, stream>>>(hsb, Wqb, bq, qb,   nullptr, M, N, K);
    gemm_bf16_wmma<0><<<gg, 256, 0, stream>>>(hsb, Wkb, bk, kbuf, nullptr, M, N, K);
    gemm_bf16_wmma<2><<<gg, 256, 0, stream>>>(hsb, Wvb, bv, vbuf, nullptr, M, N, K);

    dim3 ga(S_ / 128, H_, B_);
    attn_wmma<<<ga, 256, 0, stream>>>(qb, kbuf, vbuf, pos_bias, ctx);

    gemm_bf16_wmma<1><<<gg, 256, 0, stream>>>(ctx, Wob, bo, nullptr, out, M, N, K);
}
